// CortexModel_20040317403708
// MI455X (gfx1250) — matrix operand order repaired and checked by reading, not compiled, not run on hardware
//
#include <hip/hip_runtime.h>
#include <hip/hip_bf16.h>
#include <math.h>

// MI455X (gfx1250) implementation.
// Heavy GEMMs run on v_wmma_f32_16x16x32_bf16 (wave32, 16x16 C tile, K=32 step).
// Each wave computes a 16x64 output strip (4 N-sub-tiles) so the A fragment
// (loaded fp32 + converted to bf16 in-register) is reused by 4 independent
// WMMA chains per K-step. Weights are converted once per launch to bf16 and
// transposed to [N][K] so a lane's B fragment is 16 contiguous bf16 (2x b128).
// bf16 Wf (114MB) fits the 192MB L2, so M-tile re-reads hit L2 instead of HBM.

namespace {
constexpr int RN   = 64;
constexpr int BN   = 64;
constexpr int DN   = 256;
constexpr int VN   = 32000;
constexpr int FN   = 7;
constexpr int DFFN = 1024;
constexpr int KACT = 6;
constexpr int TOPN = 4;
constexpr float NEGV   = -1e30f;
constexpr float CTAU   = 0.20333333f;
constexpr float INVSQD = 0.0625f;   // 1/sqrt(256)
}

typedef __attribute__((ext_vector_type(16))) __bf16 v16bf;
typedef __attribute__((ext_vector_type(8)))  float  v8f;

// ---------------------------------------------------------------------------
// Weight conversion: src fp32 [batch][K][N] -> dst bf16 [batch][N][K]
// ---------------------------------------------------------------------------
__global__ void to_bf16_T(const float* __restrict__ src, __bf16* __restrict__ dst,
                          int K, int N) {
  long total = (long)K * N;
  long base  = (long)blockIdx.y * total;
  for (long i = (long)blockIdx.x * blockDim.x + threadIdx.x; i < total;
       i += (long)gridDim.x * blockDim.x) {
    long k = i / N, n = i % N;
    dst[base + n * (long)K + k] = (__bf16)src[base + i];
  }
}

// ---------------------------------------------------------------------------
// WMMA GEMM: C[M,N] = epi(A[M,K]fp32 * Bt[N,K]bf16 + bias)
// Block = 4 waves; each wave owns a 16x64 strip (4 N-sub-tiles, shared A frag).
// grid.x = (M/16)*(N/256), grid.y = batch.  Requires M%16==0, N%256==0, K%32==0.
// epi: 0 = +bias(opt), 1 = gelu(x+bias), 2 = (x+bias)*scale, 3 = x + addsrc
// ---------------------------------------------------------------------------
__global__ __launch_bounds__(128)
void wmma_gemm_bf16(const float* __restrict__ A, const __bf16* __restrict__ Bt,
                    float* __restrict__ C, const float* __restrict__ bias,
                    const float* __restrict__ addsrc, float scale,
                    int M, int N, int K,
                    long sA, long sB, long sC, long sBias, long sAdd,
                    int epi) {
  const int lane  = threadIdx.x;        // 0..31
  const int wid   = threadIdx.y;        // 0..3  (64-col strip within block)
  const int ngrp  = N >> 8;             // N/256 column groups per row of tiles
  const int tm    = blockIdx.x / ngrp;
  const int ng    = blockIdx.x % ngrp;
  const int batch = blockIdx.y;
  A  += (long)batch * sA;
  Bt += (long)batch * sB;
  C  += (long)batch * sC;
  if (bias)   bias   += (long)batch * sBias;
  if (addsrc) addsrc += (long)batch * sAdd;

  const int mrow  = tm * 16 + (lane & 15);
  const int ncol0 = ng * 256 + wid * 64 + (lane & 15);
  const int kgA   = (lane >> 4) * 16;   // A: lanes 0-15 K=0..15, 16-31 K=16..31
  const int kgB   = (lane >> 4) * 16;   // B: lanes 0-15 K=0..15, 16-31 K=16..31
  const float* arow = A + (long)mrow * K;
  const __bf16* brow0 = Bt + (long)(ncol0 +  0) * K + kgB;
  const __bf16* brow1 = Bt + (long)(ncol0 + 16) * K + kgB;
  const __bf16* brow2 = Bt + (long)(ncol0 + 32) * K + kgB;
  const __bf16* brow3 = Bt + (long)(ncol0 + 48) * K + kgB;

  v8f acc0 = {0.f,0.f,0.f,0.f,0.f,0.f,0.f,0.f};
  v8f acc1 = acc0, acc2 = acc0, acc3 = acc0;

  for (int kb = 0; kb < K; kb += 32) {
    v16bf a;
#pragma unroll
    for (int j = 0; j < 8; ++j) a[j] = (__bf16)arow[kb + kgA + j];
#pragma unroll
    for (int j = 0; j < 8; ++j) a[8 + j] = (__bf16)arow[kb + 8 + kgA + j];
    v16bf b0 = *(const v16bf*)(brow0 + kb);   // 32B-aligned: K%32==0
    v16bf b1 = *(const v16bf*)(brow1 + kb);
    v16bf b2 = *(const v16bf*)(brow2 + kb);
    v16bf b3 = *(const v16bf*)(brow3 + kb);
    if (kb + 32 < K) {                         // gfx1250 global_prefetch_b8
      __builtin_prefetch(brow0 + kb + 32, 0, 1);
      __builtin_prefetch(brow2 + kb + 32, 0, 1);
    }
    acc0 = __builtin_amdgcn_wmma_f32_16x16x32_bf16(false, a, false, b0, (short)0, acc0, false, false);
    acc1 = __builtin_amdgcn_wmma_f32_16x16x32_bf16(false, a, false, b1, (short)0, acc1, false, false);
    acc2 = __builtin_amdgcn_wmma_f32_16x16x32_bf16(false, a, false, b2, (short)0, acc2, false, false);
    acc3 = __builtin_amdgcn_wmma_f32_16x16x32_bf16(false, a, false, b3, (short)0, acc3, false, false);
  }

  v8f accs[4] = {acc0, acc1, acc2, acc3};
#pragma unroll
  for (int t = 0; t < 4; ++t) {
    const int ncol = ncol0 + 16 * t;
    const float bv = bias ? bias[ncol] : 0.f;
#pragma unroll
    for (int v = 0; v < 8; ++v) {
      int cm = tm * 16 + ((lane >> 4) << 3) + v;  // C: M = v + 8*(lane/16)
      float x = accs[t][v] + bv;
      if (epi == 1)      x = 0.5f * x * (1.0f + erff(x * 0.70710678118654752f));
      else if (epi == 2) x *= scale;
      else if (epi == 3) x += addsrc[(long)cm * N + ncol];
      C[(long)cm * N + ncol] = x;
    }
  }
}

// ---------------------------------------------------------------------------
// Region k-WTA gate: sc_g[b][r] = H[r][b]·w_gate; keep top-6 (+force r=0,R-1)
// ---------------------------------------------------------------------------
__global__ __launch_bounds__(64)
void gate_mask(const float* __restrict__ H, const float* __restrict__ wg,
               float* __restrict__ mask) {
  int b = blockIdx.x, r = threadIdx.x;
  __shared__ float sc[RN];
  __shared__ float kth;
  const float* h = H + ((long)r * BN + b) * DN;
  float s = 0.f;
  for (int d = 0; d < DN; ++d) s += h[d] * wg[d];
  sc[r] = s;
  __syncthreads();
  if (r == 0) {
    float top[KACT];
    for (int i = 0; i < KACT; ++i) top[i] = -INFINITY;
    for (int j = 0; j < RN; ++j) {
      float v = sc[j];
      for (int i = 0; i < KACT; ++i)
        if (v > top[i]) {
          for (int q = KACT - 1; q > i; --q) top[q] = top[q - 1];
          top[i] = v;
          break;
        }
    }
    kth = top[KACT - 1];
  }
  __syncthreads();
  bool on = (sc[r] >= kth) || (r == 0) || (r == RN - 1);
  mask[b * RN + r] = on ? 1.f : 0.f;
}

__global__ void hs_kernel(const float* __restrict__ H, const float* __restrict__ mask,
                          float* __restrict__ Hs) {
  long i = (long)blockIdx.x * 256 + threadIdx.x;
  if (i >= (long)RN * BN * DN) return;
  int r = (int)(i / ((long)BN * DN));
  int b = (int)((i / DN) % BN);
  Hs[i] = H[i] * mask[b * RN + r];
}

// ---------------------------------------------------------------------------
// Hex top-4 gated attention: one wave per (r,b).
// sc[k] = Q[r,b]·Vsrc[nbr[r,k],b]/sqrt(D); softmax over top-4; out = sum w*Vsrc
// ---------------------------------------------------------------------------
__global__ __launch_bounds__(32)
void msg_pass(const float* __restrict__ Q, const float* __restrict__ Vsrc,
              const int* __restrict__ nbr, float* __restrict__ outm) {
  int id = blockIdx.x, r = id / BN, b = id % BN;
  int lane = threadIdx.x;
  const float* q = Q + ((long)r * BN + b) * DN;
  int nb[6];
  float sc[6];
#pragma unroll
  for (int k = 0; k < 6; ++k) nb[k] = nbr[r * 6 + k];
#pragma unroll
  for (int k = 0; k < 6; ++k) {
    const float* vr = Vsrc + ((long)nb[k] * BN + b) * DN;
    float p = 0.f;
#pragma unroll
    for (int j = 0; j < DN / 32; ++j) { int d = lane + 32 * j; p += q[d] * vr[d]; }
#pragma unroll
    for (int o = 16; o; o >>= 1) p += __shfl_xor(p, o, 32);
    sc[k] = p * INVSQD;
  }
  float t[6];
#pragma unroll
  for (int k = 0; k < 6; ++k) t[k] = sc[k];
  for (int i = 0; i < TOPN; ++i) {            // partial selection sort
    int mi = i;
    for (int j = i + 1; j < 6; ++j) if (t[j] > t[mi]) mi = j;
    float tmp = t[i]; t[i] = t[mi]; t[mi] = tmp;
  }
  float kth = t[TOPN - 1], mx = t[0];
  float w[6], den = 0.f;
#pragma unroll
  for (int k = 0; k < 6; ++k) {
    w[k] = (sc[k] >= kth) ? __expf(sc[k] - mx) : 0.f;
    den += w[k];
  }
  float inv = 1.f / den;
#pragma unroll
  for (int j = 0; j < DN / 32; ++j) {
    int d = lane + 32 * j;
    float a = 0.f;
#pragma unroll
    for (int k = 0; k < 6; ++k)
      a += (w[k] * inv) * Vsrc[((long)nb[k] * BN + b) * DN + d];
    outm[((long)r * BN + b) * DN + d] = a;
  }
}

// ---------------------------------------------------------------------------
// Facet gate: g = X[row]·Wg  (F=7); top-2 mask; mode 0 -> log_softmax,
// mode 1 -> softmax. One wave per row.
// ---------------------------------------------------------------------------
__global__ __launch_bounds__(32)
void facet_gate(const float* __restrict__ X, const float* __restrict__ Wg,
                float* __restrict__ outv, int mode) {
  int row = blockIdx.x, lane = threadIdx.x;
  const float* x = X + (long)row * DN;
  float g[FN];
#pragma unroll
  for (int f = 0; f < FN; ++f) {
    float p = 0.f;
#pragma unroll
    for (int j = 0; j < DN / 32; ++j) { int d = lane + 32 * j; p += x[d] * Wg[d * FN + f]; }
#pragma unroll
    for (int o = 16; o; o >>= 1) p += __shfl_xor(p, o, 32);
    g[f] = p;
  }
  float m1 = -INFINITY, m2 = -INFINITY;
#pragma unroll
  for (int f = 0; f < FN; ++f) {
    float v = g[f];
    if (v > m1) { m2 = m1; m1 = v; } else if (v > m2) m2 = v;
  }
  float kth = m2, den = 0.f;
  float e[FN];
#pragma unroll
  for (int f = 0; f < FN; ++f) {
    e[f] = (g[f] >= kth) ? __expf(g[f] - m1) : 0.f;
    den += e[f];
  }
  if (lane < FN) {
    if (mode == 0)
      outv[row * FN + lane] = (g[lane] >= kth) ? (g[lane] - m1 - __logf(den)) : NEGV;
    else
      outv[row * FN + lane] = e[lane] / den;
  }
}

// ---------------------------------------------------------------------------
// Per-(f,b) logsumexp over V=32000
// ---------------------------------------------------------------------------
__global__ __launch_bounds__(256)
void row_lse(const float* __restrict__ logits, float* __restrict__ lse) {
  int fb = blockIdx.x, tid = threadIdx.x;
  const float* row = logits + (long)fb * VN;
  __shared__ float red[256];
  float mx = -INFINITY;
  for (int i = tid; i < VN; i += 256) mx = fmaxf(mx, row[i]);
  red[tid] = mx;
  __syncthreads();
  for (int s = 128; s; s >>= 1) { if (tid < s) red[tid] = fmaxf(red[tid], red[tid + s]); __syncthreads(); }
  mx = red[0];
  __syncthreads();
  float sum = 0.f;
  for (int i = tid; i < VN; i += 256) sum += __expf(row[i] - mx);
  red[tid] = sum;
  __syncthreads();
  for (int s = 128; s; s >>= 1) { if (tid < s) red[tid] += red[tid + s]; __syncthreads(); }
  if (tid == 0) lse[fb] = mx + __logf(red[0]);
}

// logp[b,v] = logsumexp_f( log_pi[b,f] + logits[f,b,v] - lse[f,b] )
__global__ void logp_kernel(const float* __restrict__ logits, const float* __restrict__ lse,
                            const float* __restrict__ logpi, float* __restrict__ outv) {
  long i = (long)blockIdx.x * 256 + threadIdx.x;
  if (i >= (long)BN * VN) return;
  int b = (int)(i / VN), v = (int)(i % VN);
  float t[FN], m = -INFINITY;
#pragma unroll
  for (int f = 0; f < FN; ++f) {
    t[f] = logpi[b * FN + f] + logits[((long)f * BN + b) * VN + v] - lse[f * BN + b];
    m = fmaxf(m, t[f]);
  }
  float s = 0.f;
#pragma unroll
  for (int f = 0; f < FN; ++f) s += __expf(t[f] - m);
  outv[i] = m + __logf(s);
}

__global__ __launch_bounds__(64)
void loss_kernel(const float* __restrict__ logp, const int* __restrict__ targets,
                 float* __restrict__ loss_out, float* __restrict__ Sout) {
  int b = threadIdx.x;
  __shared__ float red[BN];
  red[b] = logp[(long)b * VN + targets[b]];
  __syncthreads();
  for (int s = 32; s; s >>= 1) { if (b < s) red[b] += red[b + s]; __syncthreads(); }
  if (b == 0) { float L = -red[0] / BN; loss_out[0] = L; Sout[0] = L; }
}

// Bc = H2 * (pi @ e_fac) * boost(S)
__global__ void bc_kernel(const float* __restrict__ H2, const float* __restrict__ pi,
                          const float* __restrict__ efac, const float* __restrict__ Sv,
                          float* __restrict__ Bc) {
  long i = (long)blockIdx.x * 256 + threadIdx.x;
  if (i >= (long)RN * BN * DN) return;
  long rb = i / DN;
  int d = (int)(i % DN);
  float boost = 1.f + 2.f * (Sv[0] > 0.7f ? 1.f : 0.f);
  float e = 0.f;
#pragma unroll
  for (int f = 0; f < FN; ++f) e += pi[rb * FN + f] * efac[f * DN + d];
  Bc[i] = H2[i] * e * boost;
}

// ---------------------------------------------------------------------------
extern "C" void kernel_launch(void* const* d_in, const int* in_sizes, int n_in,
                              void* d_out, int out_size, void* d_ws, size_t ws_size,
                              hipStream_t stream) {
  const float* x    = (const float*)d_in[0];
  const int*   tgt  = (const int*)  d_in[1];
  const int*   nbr  = (const int*)  d_in[2];
  const float* W1   = (const float*)d_in[3];
  const float* b1   = (const float*)d_in[4];
  const float* W2   = (const float*)d_in[5];
  const float* b2   = (const float*)d_in[6];
  const float* wg   = (const float*)d_in[7];
  const float* Wr   = (const float*)d_in[8];
  const float* Wm   = (const float*)d_in[9];
  const float* bm   = (const float*)d_in[10];
  const float* Wg   = (const float*)d_in[11];
  const float* Wf   = (const float*)d_in[12];
  const float* efac = (const float*)d_in[13];
  const float* Wb   = (const float*)d_in[14];
  float* out = (float*)d_out;

  char* ws = (char*)d_ws;
  size_t off = 0;
  auto alloc = [&](size_t bytes) -> char* {
    char* p = ws + off;
    off += (bytes + 255) & ~(size_t)255;
    return p;
  };
  __bf16* W1t   = (__bf16*)alloc((size_t)RN * DFFN * DN * 2);
  __bf16* W2t   = (__bf16*)alloc((size_t)RN * DN * DFFN * 2);
  __bf16* Wrt   = (__bf16*)alloc((size_t)DN * DN * 2);
  __bf16* Wmt   = (__bf16*)alloc((size_t)DN * DN * 2);
  __bf16* Wbt   = (__bf16*)alloc((size_t)DN * DN * 2);
  __bf16* Wft   = (__bf16*)alloc((size_t)FN * VN * DN * 2);
  float* hid    = (float*)alloc((size_t)RN * BN * DFFN * 4);
  float* H      = (float*)alloc((size_t)RN * BN * DN * 4);
  float* maskb  = (float*)alloc((size_t)BN * RN * 4);
  float* Hs     = (float*)alloc((size_t)RN * BN * DN * 4);
  float* msg    = (float*)alloc((size_t)RN * BN * DN * 4);
  float* H2     = (float*)alloc((size_t)RN * BN * DN * 4);
  float* motor  = (float*)alloc((size_t)BN * DN * 4);
  float* logpi  = (float*)alloc((size_t)BN * FN * 4);
  float* logits = (float*)alloc((size_t)FN * BN * VN * 4);
  float* lse    = (float*)alloc((size_t)FN * BN * 4);
  float* Sval   = (float*)alloc(256);
  float* pi2    = (float*)alloc((size_t)RN * BN * FN * 4);
  float* Bcb    = (float*)alloc((size_t)RN * BN * DN * 4);
  float* msg2   = (float*)alloc((size_t)RN * BN * DN * 4);

  const dim3 gblk(32, 4);   // 4 waves per block, one 16x64 strip each

  // --- weights -> bf16, transposed [N][K]
  to_bf16_T<<<dim3(256, RN), 256, 0, stream>>>(W1, W1t, DN, DFFN);
  to_bf16_T<<<dim3(256, RN), 256, 0, stream>>>(W2, W2t, DFFN, DN);
  to_bf16_T<<<dim3(64, 1),   256, 0, stream>>>(Wr, Wrt, DN, DN);
  to_bf16_T<<<dim3(64, 1),   256, 0, stream>>>(Wm, Wmt, DN, DN);
  to_bf16_T<<<dim3(64, 1),   256, 0, stream>>>(Wb, Wbt, DN, DN);
  to_bf16_T<<<dim3(4096, FN), 256, 0, stream>>>(Wf, Wft, DN, VN);

  // --- per-region FF: hid = gelu(x@W1 + b1); H = (1+CTAU)*(hid@W2 + b2)
  wmma_gemm_bf16<<<dim3((BN / 16) * (DFFN / 256), RN), gblk, 0, stream>>>(
      x, W1t, hid, b1, nullptr, 1.f, BN, DFFN, DN,
      (long)BN * DN, (long)DFFN * DN, (long)BN * DFFN, DFFN, 0, 1);
  wmma_gemm_bf16<<<dim3((BN / 16) * (DN / 256), RN), gblk, 0, stream>>>(
      hid, W2t, H, b2, nullptr, 1.f + CTAU, BN, DN, DFFN,
      (long)BN * DFFN, (long)DN * DFFN, (long)BN * DN, DN, 0, 2);

  // --- region k-WTA gate, Hs = H*mask
  gate_mask<<<BN, RN, 0, stream>>>(H, wg, maskb);
  hs_kernel<<<(RN * BN * DN + 255) / 256, 256, 0, stream>>>(H, maskb, Hs);

  // --- message pass 1, H2 = Hs + msg@Wr
  msg_pass<<<RN * BN, 32, 0, stream>>>(Hs, Hs, nbr, msg);
  wmma_gemm_bf16<<<dim3((RN * BN / 16) * (DN / 256), 1), gblk, 0, stream>>>(
      msg, Wrt, H2, nullptr, Hs, 1.f, RN * BN, DN, DN, 0, 0, 0, 0, 0, 3);

  // --- motor readout + facet mixture over vocab
  wmma_gemm_bf16<<<dim3((BN / 16) * (DN / 256), 1), gblk, 0, stream>>>(
      H2 + (long)(RN - 1) * BN * DN, Wmt, motor, bm, nullptr, 1.f,
      BN, DN, DN, 0, 0, 0, 0, 0, 0);
  facet_gate<<<BN, 32, 0, stream>>>(motor, Wg, logpi, 0);
  wmma_gemm_bf16<<<dim3((BN / 16) * (VN / 256), FN), gblk, 0, stream>>>(
      motor, Wft, logits, nullptr, nullptr, 1.f, BN, VN, DN,
      0, (long)VN * DN, (long)BN * VN, 0, 0, 0);
  row_lse<<<FN * BN, 256, 0, stream>>>(logits, lse);
  logp_kernel<<<(BN * VN + 255) / 256, 256, 0, stream>>>(logits, lse, logpi, out);
  loss_kernel<<<1, BN, 0, stream>>>(out, tgt, out + (size_t)BN * VN, Sval);

  // --- surprise-gated emitter + broadcast routing
  facet_gate<<<RN * BN, 32, 0, stream>>>(H2, Wg, pi2, 1);
  bc_kernel<<<(RN * BN * DN + 255) / 256, 256, 0, stream>>>(H2, pi2, efac, Sval, Bcb);
  msg_pass<<<RN * BN, 32, 0, stream>>>(H2, Bcb, nbr, msg2);
  wmma_gemm_bf16<<<dim3((RN * BN / 16) * (DN / 256), 1), gblk, 0, stream>>>(
      msg2, Wbt, out + (size_t)BN * VN + 1, nullptr, nullptr, 1.f,
      RN * BN, DN, DN, 0, 0, 0, 0, 0, 0);

  (void)in_sizes; (void)n_in; (void)out_size; (void)ws_size;
}
